// FusedMoE_76905684402346
// MI455X (gfx1250) — matrix operand order repaired and checked by reading, not compiled, not run on hardware
//
#include <hip/hip_runtime.h>
#include <hip/hip_bf16.h>

#define NE   8
#define HID  2048
#define ITR  4096
#define NT   4096

#define BM   256         // token rows per block (both GEMMs)
#define BN   128         // output cols per block (GEMM1: 128 up + 128 gate)
#define KC   64
#define SAP  (KC + 8)    // padded LDS row stride in bf16 elements (144B, 16B-multiple)
#define NTHR 512         // 16 waves

typedef __attribute__((ext_vector_type(16))) __bf16 v16bf;
typedef __attribute__((ext_vector_type(8)))  __bf16 v8bf;
typedef __attribute__((ext_vector_type(4)))  __bf16 v4bf;
typedef __attribute__((ext_vector_type(8)))  float  v8f;

// ---------------- fragment loaders (LDS bf16, row-major, stride SAP) -------
// A (16x32): lane holds row M=lane%16; K elems 0-7 & 16-23 (lanes 0-15),
//            8-15 & 24-31 (lanes 16-31).
__device__ __forceinline__ v16bf ld_fragA(const __bf16* base, int lane) {
  const __bf16* p = base + (lane & 15) * SAP + ((lane >> 4) << 3);
  v8bf lo = *(const v8bf*)(p);
  v8bf hi = *(const v8bf*)(p + 16);
  return __builtin_shufflevector(lo, hi, 0,1,2,3,4,5,6,7,8,9,10,11,12,13,14,15);
}
// B (32x16): lane holds col N=lane%16 (a row of W, since B = W^T);
//            K elems 0-7 & 16-23 (lanes 0-15), 8-15 & 24-31 (lanes 16-31).
__device__ __forceinline__ v16bf ld_fragB(const __bf16* base, int lane) {
  const __bf16* p = base + (lane & 15) * SAP + ((lane >> 4) << 3);
  v8bf lo = *(const v8bf*)(p);
  v8bf hi = *(const v8bf*)(p + 16);
  return __builtin_shufflevector(lo, hi, 0,1,2,3,4,5,6,7,8,9,10,11,12,13,14,15);
}

__device__ __forceinline__ void cvt_store4(__bf16* d, float4 v) {
  v4bf o;
  o[0] = (__bf16)v.x; o[1] = (__bf16)v.y; o[2] = (__bf16)v.z; o[3] = (__bf16)v.w;
  *(v4bf*)d = o;
}

// async global->LDS 16B copy (CDNA5, ASYNCcnt-tracked, bypasses VGPRs)
__device__ __forceinline__ void async_copy16(void* lds_dst, const void* gsrc) {
  unsigned ldsa = (unsigned)(unsigned long long)lds_dst;   // low 32b = LDS byte addr
  asm volatile("global_load_async_to_lds_b128 %0, %1, off"
               :: "v"(ldsa), "v"(gsrc) : "memory");
}
__device__ __forceinline__ void wait_async0() {
  asm volatile("s_wait_asynccnt 0x0" ::: "memory");
}

// ---------------- router: top-2 softmax + per-expert gather lists ----------
__global__ void moe_router(const float* __restrict__ logits,
                           int* __restrict__ cnt, int* __restrict__ tok,
                           int* __restrict__ prow, float* __restrict__ wgt) {
  int t = blockIdx.x * blockDim.x + threadIdx.x;
  if (t >= NT) return;
  float l[NE];
#pragma unroll
  for (int e = 0; e < NE; ++e) l[e] = logits[t * NE + e];
  int i1 = 0; float m1 = l[0];
#pragma unroll
  for (int e = 1; e < NE; ++e) if (l[e] > m1) { m1 = l[e]; i1 = e; }
  int i2 = -1; float m2 = -3.4e38f;
#pragma unroll
  for (int e = 0; e < NE; ++e) if (e != i1 && l[e] > m2) { m2 = l[e]; i2 = e; }
  float w1 = 1.0f / (1.0f + __expf(m2 - m1));   // normalized top-2 softmax
  float w2 = 1.0f - w1;
  int s1 = atomicAdd(&cnt[i1], 1);
  tok[i1 * NT + s1] = t; prow[i1 * NT + s1] = 2 * t;     wgt[i1 * NT + s1] = w1;
  int s2 = atomicAdd(&cnt[i2], 1);
  tok[i2 * NT + s2] = t; prow[i2 * NT + s2] = 2 * t + 1; wgt[i2 * NT + s2] = w2;
}

// ---------------- GEMM1: inter = silu(x@Wg^T) * (x@Wu^T), bf16 out ---------
__global__ void __launch_bounds__(NTHR)
moe_gemm1(const float* __restrict__ x, const float* __restrict__ w31,
          const int* __restrict__ cnt, const int* __restrict__ tok,
          const int* __restrict__ prow, __bf16* __restrict__ inter) {
  const int e = blockIdx.z;
  const int nrows = cnt[e];
  const int rb = blockIdx.y * BM;
  if (rb >= nrows) return;
  const int n0 = blockIdx.x * BN;

  __shared__ __align__(16) __bf16 sA[BM * SAP];   // 256 x 64 bf16
  __shared__ __align__(16) __bf16 sU[BN * SAP];   // 128 x 64
  __shared__ __align__(16) __bf16 sG[BN * SAP];
  __shared__ int sTok[BM];
  __shared__ int sPr[BM];

  const int tid  = threadIdx.x;
  const int lane = tid & 31;
  const int wm   = (tid >> 5) >> 2;   // 0..3 -> 64 rows each
  const int wn   = (tid >> 5) & 3;    // 0..3 -> 32 up + 32 gate cols each

  if (tid < BM) {
    int r = rb + tid;
    int rv = (r < nrows) ? r : (nrows - 1);
    sTok[tid] = tok[e * NT + rv];
    sPr[tid]  = (r < nrows) ? prow[e * NT + r] : -1;
  }
  __syncthreads();

  v8f accU[4][2], accG[4][2];
#pragma unroll
  for (int i = 0; i < 4; ++i)
#pragma unroll
    for (int j = 0; j < 2; ++j) { accU[i][j] = (v8f)(0.0f); accG[i][j] = (v8f)(0.0f); }

  const float* wb = w31 + (size_t)e * (2 * ITR) * HID;

  for (int k0 = 0; k0 < HID; k0 += KC) {
#pragma unroll
    for (int i = 0; i < 8; ++i) {             // A: 256 rows x 16 float4 = 4096 slots
      int slot = tid + i * NTHR;
      int row  = slot >> 4;
      int c4   = (slot & 15) << 2;
      float4 v = *(const float4*)(x + (size_t)sTok[row] * HID + k0 + c4);
      cvt_store4(sA + row * SAP + c4, v);
    }
#pragma unroll
    for (int i = 0; i < 4; ++i) {             // U/G: 128 rows x 16 float4 each
      int slot = tid + i * NTHR;
      int row  = slot >> 4;
      int c4   = (slot & 15) << 2;
      const float* pu = wb + (size_t)(n0 + row) * HID + k0 + c4;
      const float* pg = wb + (size_t)(ITR + n0 + row) * HID + k0 + c4;
      float4 vu = *(const float4*)pu;
      float4 vg = *(const float4*)pg;
      cvt_store4(sU + row * SAP + c4, vu);
      cvt_store4(sG + row * SAP + c4, vg);
      if (k0 + KC < HID) {                    // prefetch next K-tile of weights
        __builtin_prefetch(pu + KC, 0, 3);
        __builtin_prefetch(pg + KC, 0, 3);
      }
    }
    __syncthreads();

#pragma unroll
    for (int kk = 0; kk < KC; kk += 32) {
      v16bf a[4], bu[2], bg[2];
#pragma unroll
      for (int rt = 0; rt < 4; ++rt)
        a[rt] = ld_fragA(sA + (wm * 64 + rt * 16) * SAP + kk, lane);
#pragma unroll
      for (int ct = 0; ct < 2; ++ct) {
        bu[ct] = ld_fragB(sU + (wn * 32 + ct * 16) * SAP + kk, lane);
        bg[ct] = ld_fragB(sG + (wn * 32 + ct * 16) * SAP + kk, lane);
      }
#pragma unroll
      for (int rt = 0; rt < 4; ++rt)
#pragma unroll
        for (int ct = 0; ct < 2; ++ct) {
          accU[rt][ct] = __builtin_amdgcn_wmma_f32_16x16x32_bf16(
              false, a[rt], false, bu[ct], (short)0, accU[rt][ct], false, false);
          accG[rt][ct] = __builtin_amdgcn_wmma_f32_16x16x32_bf16(
              false, a[rt], false, bg[ct], (short)0, accG[rt][ct], false, false);
        }
    }
    __syncthreads();
  }

  const int mhi  = (lane >> 4) << 3;
  const int ncol = lane & 15;
#pragma unroll
  for (int rt = 0; rt < 4; ++rt)
#pragma unroll
    for (int ct = 0; ct < 2; ++ct)
#pragma unroll
      for (int el = 0; el < 8; ++el) {
        int lrow = wm * 64 + rt * 16 + el + mhi;
        int pr = sPr[lrow];
        if (pr >= 0) {
          float u = accU[rt][ct][el];
          float g = accG[rt][ct][el];
          float s = g / (1.0f + __expf(-g));
          int col = n0 + wn * 32 + ct * 16 + ncol;
          inter[(size_t)pr * ITR + col] = (__bf16)(s * u);
        }
      }
}

// ---------------- GEMM2: out += cw * (inter @ w2^T), atomic scatter --------
__global__ void __launch_bounds__(NTHR)
moe_gemm2(const __bf16* __restrict__ inter, const float* __restrict__ w2,
          const int* __restrict__ cnt, const int* __restrict__ tok,
          const int* __restrict__ prow, const float* __restrict__ wgt,
          float* __restrict__ out) {
  const int e = blockIdx.z;
  const int nrows = cnt[e];
  const int rb = blockIdx.y * BM;
  if (rb >= nrows) return;
  const int n0 = blockIdx.x * BN;          // over HID

  __shared__ __align__(16) __bf16 sA[BM * SAP];   // 256 x 64 bf16 (async-filled)
  __shared__ __align__(16) __bf16 sB[BN * SAP];   // 128 x 64
  __shared__ int   sTok[BM];
  __shared__ int   sPr[BM];
  __shared__ float sW[BM];

  const int tid  = threadIdx.x;
  const int lane = tid & 31;
  const int wm   = (tid >> 5) >> 2;   // 0..3 -> 64 rows
  const int wn   = (tid >> 5) & 3;    // 0..3 -> 32 cols

  if (tid < BM) {
    int r = rb + tid;
    bool v = (r < nrows);
    sPr[tid]  = v ? prow[e * NT + r] : 0;
    sTok[tid] = v ? tok[e * NT + r] : 0;
    sW[tid]   = v ? wgt[e * NT + r] : 0.0f;
  }
  __syncthreads();

  v8f acc[4][2];
#pragma unroll
  for (int i = 0; i < 4; ++i)
#pragma unroll
    for (int j = 0; j < 2; ++j) acc[i][j] = (v8f)(0.0f);

  const float* w2b = w2 + (size_t)e * HID * ITR;

  for (int k0 = 0; k0 < ITR; k0 += KC) {
#pragma unroll
    for (int i = 0; i < 4; ++i) {            // A: 256 rows x 8 16B-chunks = 2048 slots
      int slot = tid + i * NTHR;             // async global->LDS, no VGPR staging
      int row  = slot >> 3;
      int c8   = (slot & 7) << 3;
      async_copy16(sA + row * SAP + c8,
                   inter + (size_t)sPr[row] * ITR + k0 + c8);
    }
#pragma unroll
    for (int i = 0; i < 4; ++i) {            // B: 128 rows x 16 float4, f32->bf16
      int slot = tid + i * NTHR;
      int row  = slot >> 4;
      int c4   = (slot & 15) << 2;
      const float* pb = w2b + (size_t)(n0 + row) * ITR + k0 + c4;
      float4 v = *(const float4*)pb;
      cvt_store4(sB + row * SAP + c4, v);
      if (k0 + KC < ITR) __builtin_prefetch(pb + KC, 0, 3);
    }
    wait_async0();
    __syncthreads();

#pragma unroll
    for (int kk = 0; kk < KC; kk += 32) {
      v16bf a[4], b[2];
#pragma unroll
      for (int rt = 0; rt < 4; ++rt)
        a[rt] = ld_fragA(sA + (wm * 64 + rt * 16) * SAP + kk, lane);
#pragma unroll
      for (int ct = 0; ct < 2; ++ct)
        b[ct] = ld_fragB(sB + (wn * 32 + ct * 16) * SAP + kk, lane);
#pragma unroll
      for (int rt = 0; rt < 4; ++rt)
#pragma unroll
        for (int ct = 0; ct < 2; ++ct)
          acc[rt][ct] = __builtin_amdgcn_wmma_f32_16x16x32_bf16(
              false, a[rt], false, b[ct], (short)0, acc[rt][ct], false, false);
    }
    __syncthreads();
  }

  const int mhi  = (lane >> 4) << 3;
  const int ncol = lane & 15;
#pragma unroll
  for (int rt = 0; rt < 4; ++rt)
#pragma unroll
    for (int ct = 0; ct < 2; ++ct)
#pragma unroll
      for (int el = 0; el < 8; ++el) {
        int lrow = wm * 64 + rt * 16 + el + mhi;
        float v = acc[rt][ct][el] * sW[lrow];   // 0 for padded rows
        int col = n0 + wn * 32 + ct * 16 + ncol;
        atomicAdd(&out[(size_t)sTok[lrow] * HID + col], v);
      }
}

// ---------------- launch ---------------------------------------------------
extern "C" void kernel_launch(void* const* d_in, const int* in_sizes, int n_in,
                              void* d_out, int out_size, void* d_ws, size_t ws_size,
                              hipStream_t stream) {
  (void)in_sizes; (void)n_in; (void)ws_size;
  const float* x      = (const float*)d_in[0];
  const float* logits = (const float*)d_in[1];
  const float* w31    = (const float*)d_in[2];
  const float* w2     = (const float*)d_in[3];
  float* out = (float*)d_out;

  char* ws = (char*)d_ws;
  size_t off = 0;
  int*   cnt  = (int*)(ws + off);   off += 256;
  int*   tok  = (int*)(ws + off);   off += (size_t)NE * NT * sizeof(int);
  int*   prw  = (int*)(ws + off);   off += (size_t)NE * NT * sizeof(int);
  float* wgt  = (float*)(ws + off); off += (size_t)NE * NT * sizeof(float);
  off = (off + 255) & ~(size_t)255;
  __bf16* inter = (__bf16*)(ws + off);     // [2*NT][ITR] bf16 = 64 MB

  hipMemsetAsync(cnt, 0, NE * sizeof(int), stream);
  hipMemsetAsync(out, 0, (size_t)out_size * sizeof(float), stream);

  moe_router<<<NT / 256, 256, 0, stream>>>(logits, cnt, tok, prw, wgt);
  moe_gemm1<<<dim3(ITR / BN, NT / BM, NE), NTHR, 0, stream>>>(x, w31, cnt, tok, prw, inter);
  moe_gemm2<<<dim3(HID / BN, NT / BM, NE), NTHR, 0, stream>>>(inter, w2, cnt, tok, prw, wgt, out);
}
